// speaker_TRM_18442589569060
// MI455X (gfx1250) — matrix operand order repaired and checked by reading, not compiled, not run on hardware
//
#include <hip/hip_runtime.h>
#include <hip/hip_bf16.h>
#include <math.h>

// ---------------- problem constants ----------------
constexpr int kS  = 512;
constexpr int kB  = 16;
constexpr int kD  = 512;
constexpr int kH  = 8;
constexpr int kFF = 2048;
constexpr int kL  = 2;
constexpr int kWIN = 3;
constexpr int kDH = 64;
constexpr int kNE = 4;
constexpr int kM  = kB * kS;          // 8192 rows in every GEMM
constexpr float kNEG = -1e9f;

// ---------------- CDNA5 async-copy availability ----------------
#if defined(__gfx1250__) && __has_builtin(__builtin_amdgcn_global_load_async_to_lds_b128) && __has_builtin(__builtin_amdgcn_s_wait_asynccnt)
#define ASYNC_OK 1
#else
#define ASYNC_OK 0
#endif

typedef int v4i_vec __attribute__((vector_size(16)));   // matches builtin param type

#if ASYNC_OK
__device__ __forceinline__ void async_b128(const void* g, void* l) {
  __builtin_amdgcn_global_load_async_to_lds_b128(
      (__attribute__((address_space(1))) v4i_vec*)g,
      (__attribute__((address_space(3))) v4i_vec*)l, 0, 0);
}
__device__ __forceinline__ void await_async() {
  __builtin_amdgcn_s_wait_asynccnt(0);
}
#endif

// ---------------- WMMA types ----------------
typedef __attribute__((ext_vector_type(16))) __bf16 v16bf;
typedef __attribute__((ext_vector_type(8)))  float  v8f;

union FragB { v16bf v; unsigned u[8]; };

__device__ __forceinline__ v8f wmma_bf16(v16bf a, v16bf b, v8f c) {
  return __builtin_amdgcn_wmma_f32_16x16x32_bf16(false, a, false, b, (short)0, c, false, false);
}

__device__ __forceinline__ unsigned short f2bf_bits(float x) {
  union { float f; unsigned u; } a; a.f = x;
  unsigned r = a.u + 0x7FFFu + ((a.u >> 16) & 1u);   // round-to-nearest-even
  return (unsigned short)(r >> 16);
}

// =====================================================================
// GEMM: C(M,N) = A(M,K) * W(N,K)^T + bias ; A,W bf16, accum fp32 (WMMA)
// mode 0: bf16 out ; mode 1: bf16 relu out ; mode 2: fp32 out
// Block 256 thr = 8 waves; tile 128x128; K-step 32.
// Async path: double-buffered GLOBAL_LOAD_ASYNC_TO_LDS_B128 pipeline
// (ASYNCcnt), one barrier per K-step, copies overlap the WMMA phase.
// =====================================================================
constexpr int kBM = 128, kBN = 128, kBK = 32;
constexpr int kLDTD = 20;   // dwords per LDS tile row (32 bf16 + pad); 80B keeps
                            // every per-lane async B128 dest 16-byte aligned

__global__ __launch_bounds__(256)
void gemm_bf16_kernel(const unsigned short* __restrict__ A,
                      const unsigned short* __restrict__ Wt,
                      const float* __restrict__ bias,
                      unsigned short* __restrict__ outb,
                      float* __restrict__ outf,
                      int Mtot, int Ntot, int Ktot, int mode) {
#if ASYNC_OK
  __shared__ unsigned sA[2][kBM * kLDTD];
  __shared__ unsigned sB[2][kBN * kLDTD];
#else
  __shared__ unsigned sA[1][kBM * kLDTD];
  __shared__ unsigned sB[1][kBN * kLDTD];
#endif

  const int t     = threadIdx.x;
  const int w     = t >> 5;
  const int lane  = t & 31;
  const int bm    = blockIdx.y * kBM;
  const int bn    = blockIdx.x * kBN;
  const int waveM = (w & 3) * 32;   // 4 waves along M
  const int waveN = (w >> 2) * 64;  // 2 waves along N
  const int mlane = lane & 15;
  const int khalf = lane >> 4;

  v8f acc[2][4];
#pragma unroll
  for (int i = 0; i < 2; i++)
#pragma unroll
    for (int j = 0; j < 4; j++)
      acc[i][j] = (v8f){0.f,0.f,0.f,0.f,0.f,0.f,0.f,0.f};

  const int lrow = t >> 1;          // 128 rows, 2 threads/row
  const int lcol = (t & 1) * 16;    // 16 bf16 per thread per tile row

  // fragment loads + 8 WMMAs for one K-step from given LDS buffers
  auto compute_step = [&](const unsigned* __restrict__ cA,
                          const unsigned* __restrict__ cB) {
    FragB af[2];
#pragma unroll
    for (int mt = 0; mt < 2; mt++) {
      const unsigned* base = cA + (waveM + mt * 16 + mlane) * kLDTD;
#pragma unroll
      for (int j = 0; j < 8; j++) {
        int k = (j < 4 ? 2 * j : 16 + 2 * (j - 4)) + khalf * 8;
        af[mt].u[j] = base[k >> 1];
      }
    }
#pragma unroll
    for (int nt = 0; nt < 4; nt++) {
      FragB bf;   // B layout: lane = N col, lanes 0-15 K=0..7,16..23, lanes 16-31 K=8..15,24..31
      const unsigned* base = cB + (waveN + nt * 16 + mlane) * kLDTD;
#pragma unroll
      for (int j = 0; j < 8; j++) {
        int k = (j < 4 ? 2 * j : 16 + 2 * (j - 4)) + khalf * 8;
        bf.u[j] = base[k >> 1];
      }
#pragma unroll
      for (int mt = 0; mt < 2; mt++)
        acc[mt][nt] = wmma_bf16(af[mt].v, bf.v, acc[mt][nt]);
    }
  };

#if ASYNC_OK
  // issue async tile copies for K-step k0 into buffer `buf`
  auto issue = [&](int k0, int buf) {
    const unsigned short* ga = A  + (size_t)(bm + lrow) * Ktot + k0 + lcol;
    const unsigned short* gb = Wt + (size_t)(bn + lrow) * Ktot + k0 + lcol;
    unsigned* la = &sA[buf][lrow * kLDTD + (t & 1) * 8];
    unsigned* lb = &sB[buf][lrow * kLDTD + (t & 1) * 8];
    async_b128(ga,     la);
    async_b128(ga + 8, la + 4);
    async_b128(gb,     lb);
    async_b128(gb + 8, lb + 4);
  };
  issue(0, 0);
  int buf = 0;
  for (int k0 = 0; k0 < Ktot; k0 += kBK) {
    await_async();                       // own copies done
    __syncthreads();                     // everyone's copies done
    if (k0 + kBK < Ktot) issue(k0 + kBK, buf ^ 1);   // prefetch next tile
    compute_step(sA[buf], sB[buf]);
    buf ^= 1;
  }
#else
  for (int k0 = 0; k0 < Ktot; k0 += kBK) {
    const uint4* gA = (const uint4*)(A  + (size_t)(bm + lrow) * Ktot + k0 + lcol);
    const uint4* gB = (const uint4*)(Wt + (size_t)(bn + lrow) * Ktot + k0 + lcol);
    uint4 a0 = gA[0], a1 = gA[1];
    uint4 b0 = gB[0], b1 = gB[1];
    if (k0 + kBK < Ktot) {
      __builtin_prefetch(A  + (size_t)(bm + lrow) * Ktot + k0 + kBK + lcol, 0, 3);
      __builtin_prefetch(Wt + (size_t)(bn + lrow) * Ktot + k0 + kBK + lcol, 0, 3);
    }
    __syncthreads();
    unsigned* wA = sA[0] + lrow * kLDTD + (t & 1) * 8;
    wA[0]=a0.x; wA[1]=a0.y; wA[2]=a0.z; wA[3]=a0.w;
    wA[4]=a1.x; wA[5]=a1.y; wA[6]=a1.z; wA[7]=a1.w;
    unsigned* wB = sB[0] + lrow * kLDTD + (t & 1) * 8;
    wB[0]=b0.x; wB[1]=b0.y; wB[2]=b0.z; wB[3]=b0.w;
    wB[4]=b1.x; wB[5]=b1.y; wB[6]=b1.z; wB[7]=b1.w;
    __syncthreads();
    compute_step(sA[0], sB[0]);
  }
#endif

  // store: C/D layout — VGPR i: lanes0-15 M=i, lanes16-31 M=8+i; lane&15 = N
#pragma unroll
  for (int mt = 0; mt < 2; mt++)
#pragma unroll
    for (int nt = 0; nt < 4; nt++) {
      int col = bn + waveN + nt * 16 + mlane;
      float bv = bias ? bias[col] : 0.f;
#pragma unroll
      for (int i = 0; i < 8; i++) {
        int row = bm + waveM + mt * 16 + khalf * 8 + i;
        float y = acc[mt][nt][i] + bv;
        size_t idx = (size_t)row * Ntot + col;
        if (mode == 2)      outf[idx] = y;
        else if (mode == 1) outb[idx] = f2bf_bits(fmaxf(y, 0.f));
        else                outb[idx] = f2bf_bits(y);
      }
    }
}

// =====================================================================
// Fused attention: per block = (b, h, 32-query tile).
// Stage0: async-stage whole V head-tile (64KB) into LDS (ASYNCcnt) —
//         overlaps with score WMMAs + softmax, waited before stage 3.
// Stage1: scores (32x512) via WMMA -> LDS fp32
// Stage2: masked softmax in fp32 (bias computed on the fly per expert)
// Stage3: O = P(32x512) x V(512x64) via WMMA from LDS
// Uses ~162KB dynamic LDS — CDNA5's 320KB WGP LDS makes this legal.
// =====================================================================
constexpr int kSROW  = 520;  // padded score row stride (floats / bf16 elems)
constexpr int kVOFF  = 32 * kSROW * 4 + 32 * kSROW * 2;       // 99840 (16B aligned)
constexpr int kATTN_SH = kVOFF + kS * kDH * 2;                // + 64KB V tile

__global__ __launch_bounds__(256)
void attn_kernel(const unsigned short* __restrict__ qkv,  // (B,S,3D) bf16
                 const int* __restrict__ spk,             // (S,B) int32
                 const unsigned char* __restrict__ kpm,   // (B,S) bool
                 unsigned short* __restrict__ obuf,       // (B,S,D) bf16
                 int expert) {
  extern __shared__ unsigned char smraw[];
  float*          sc   = (float*)smraw;                             // 32 x kSROW fp32
  unsigned short* pbuf = (unsigned short*)(smraw + 32 * kSROW * 4); // 32 x kSROW bf16
  unsigned short* sV   = (unsigned short*)(smraw + kVOFF);          // 512 x 64 bf16
  __shared__ float redmax[32][8];
  __shared__ float redsum[32][8];

  const int t    = threadIdx.x;
  const int w    = t >> 5;
  const int lane = t & 31;
  const int blk  = blockIdx.x;         // b*128 + h*16 + qt
  const int qt   = blk & 15;
  const int h    = (blk >> 4) & 7;
  const int b    = blk >> 7;
  const int q0   = qt * 32;

  const size_t rowstride = 3 * kD;     // 1536 bf16 per (b,s) row of qkv
  const size_t hoff = (size_t)b * kS * rowstride + h * kDH;
  const unsigned short* Qp = qkv + hoff;
  const unsigned short* Kp = qkv + hoff + kD;
  const unsigned short* Vp = qkv + hoff + 2 * kD;

  const int mlane = lane & 15;
  const int khalf = lane >> 4;
  const float scale = 0.125f;          // 1/sqrt(64)

#if ASYNC_OK
  // ---------- stage 0: kick off async V staging (512 x 64 bf16) ----------
#pragma unroll 4
  for (int i = 0; i < 16; i++) {
    int idx = t + i * 256;             // 4096 chunks of 8 bf16 (16B)
    int row = idx >> 3, seg = idx & 7;
    async_b128(Vp + (size_t)row * rowstride + seg * 8, sV + row * kDH + seg * 8);
  }
#endif

  // ---------- stage 1: scores = Q K^T * scale ----------
  FragB qa[2][2];                       // [m-tile][k-step], reused across all N tiles
#pragma unroll
  for (int mt = 0; mt < 2; mt++) {
    const unsigned short* qrow = Qp + (size_t)(q0 + mt * 16 + mlane) * rowstride;
#pragma unroll
    for (int ks = 0; ks < 2; ks++)
#pragma unroll
      for (int j = 0; j < 8; j++) {
        int d = (j < 4 ? 2 * j : 16 + 2 * (j - 4)) + khalf * 8 + ks * 32;
        qa[mt][ks].u[j] = *(const unsigned*)(qrow + d);
      }
  }
#pragma unroll
  for (int nt = 0; nt < 4; nt++) {
    v8f acc0 = (v8f){0.f,0.f,0.f,0.f,0.f,0.f,0.f,0.f};
    v8f acc1 = (v8f){0.f,0.f,0.f,0.f,0.f,0.f,0.f,0.f};
    const unsigned short* krow = Kp + (size_t)(w * 64 + nt * 16 + mlane) * rowstride;
#pragma unroll
    for (int ks = 0; ks < 2; ks++) {
      FragB kb;                        // B = K^T: lane = key col, j pairs along d
#pragma unroll
      for (int j = 0; j < 8; j++) {
        int d = (j < 4 ? 2 * j : 16 + 2 * (j - 4)) + khalf * 8 + ks * 32;
        kb.u[j] = *(const unsigned*)(krow + d);
      }
      acc0 = wmma_bf16(qa[0][ks].v, kb.v, acc0);
      acc1 = wmma_bf16(qa[1][ks].v, kb.v, acc1);
    }
#pragma unroll
    for (int i = 0; i < 8; i++) {
      int c = w * 64 + nt * 16 + mlane;
      sc[(khalf * 8 + i) * kSROW + c]      = acc0[i] * scale;
      sc[(16 + khalf * 8 + i) * kSROW + c] = acc1[i] * scale;
    }
  }
  __syncthreads();

  // ---------- stage 2: masked softmax (8 threads per row) ----------
  const int r  = t >> 3;
  const int sg = t & 7;
  const int c0 = sg * 64;
  const int qg = q0 + r;
  const int spq = spk[(size_t)qg * kB + b];
  float pmax = -3.4e38f;
  for (int c = c0; c < c0 + 64; c++) {
    float v = sc[r * kSROW + c];
    float bias = kpm[(size_t)b * kS + c] ? kNEG : 0.f;
    int spc = spk[(size_t)c * kB + b];
    if (expert == 1)      { if (spc != spq) bias += kNEG; }
    else if (expert == 2) { if (spc == spq && c != qg) bias += kNEG; }
    else if (expert == 3) { int dd = c - qg; if (dd < 0) dd = -dd; if (dd >= kWIN) bias += kNEG; }
    v += bias;
    sc[r * kSROW + c] = v;
    pmax = fmaxf(pmax, v);
  }
  redmax[r][sg] = pmax;
  __syncthreads();
  float mx = redmax[r][0];
#pragma unroll
  for (int i = 1; i < 8; i++) mx = fmaxf(mx, redmax[r][i]);
  float psum = 0.f;
  for (int c = c0; c < c0 + 64; c++) {
    float e = __expf(sc[r * kSROW + c] - mx);
    sc[r * kSROW + c] = e;
    psum += e;
  }
  redsum[r][sg] = psum;
  __syncthreads();
  float ssum = 0.f;
#pragma unroll
  for (int i = 0; i < 8; i++) ssum += redsum[r][i];
  float inv = 1.0f / ssum;
  for (int c = c0; c < c0 + 64; c++)
    pbuf[r * kSROW + c] = f2bf_bits(sc[r * kSROW + c] * inv);

#if ASYNC_OK
  await_async();     // own V chunks landed; barrier below covers other waves'
#endif
  __syncthreads();

  // ---------- stage 3: O = P x V (one 16x16 tile per wave) ----------
  const int mt = w & 1;
  const int nt = w >> 1;                       // 4 tiles cover d = 0..63
  const int dcol = nt * 16 + mlane;
  v8f oacc = (v8f){0.f,0.f,0.f,0.f,0.f,0.f,0.f,0.f};
  const unsigned* Pd = (const unsigned*)pbuf;  // dword row stride = kSROW/2
  const unsigned* prow = Pd + (size_t)(mt * 16 + mlane) * (kSROW / 2);
#pragma unroll 4
  for (int ks = 0; ks < 16; ks++) {
    FragB pa, vb;
#pragma unroll
    for (int j = 0; j < 8; j++) {
      int kk = (j < 4 ? 2 * j : 16 + 2 * (j - 4)) + khalf * 8 + ks * 32;
      pa.u[j] = prow[kk >> 1];
    }
#pragma unroll
    for (int j = 0; j < 8; j++) {              // B = V: lane = d col, k pairs down rows
      int k = ks * 32 + (j < 4 ? 2 * j : 16 + 2 * (j - 4)) + khalf * 8;
#if ASYNC_OK
      unsigned lo = sV[(size_t)k * kDH + dcol];
      unsigned hi = sV[(size_t)(k + 1) * kDH + dcol];
#else
      const unsigned short* v0 = Vp + (size_t)k * rowstride + dcol;
      unsigned lo = *v0;
      unsigned hi = *(v0 + rowstride);
#endif
      vb.u[j] = lo | (hi << 16);
    }
    oacc = wmma_bf16(pa.v, vb.v, oacc);
  }
#pragma unroll
  for (int i = 0; i < 8; i++) {
    int row = q0 + mt * 16 + khalf * 8 + i;
    int col = h * kDH + dcol;
    obuf[(size_t)(b * kS + row) * kD + col] = f2bf_bits(oacc[i]);
  }
}

// =====================================================================
// Residual + LayerNorm (one block per row of 512)
// accmode: 0 none, 1 store to accum, 2 add to accum
// =====================================================================
__global__ __launch_bounds__(128)
void ln_kernel(const float* __restrict__ xres, const float* __restrict__ yadd,
               const float* __restrict__ gamma, const float* __restrict__ beta,
               float* __restrict__ outf, unsigned short* __restrict__ outb,
               float* __restrict__ accum, int accmode) {
  __shared__ float red[128];
  __shared__ float s_mean, s_rstd;
  const int row = blockIdx.x;
  const int t = threadIdx.x;
  const size_t base = (size_t)row * kD;
  float v[4];
  float lsum = 0.f;
#pragma unroll
  for (int j = 0; j < 4; j++) {
    int d = t + j * 128;
    float x = xres[base + d] + (yadd ? yadd[base + d] : 0.f);
    v[j] = x; lsum += x;
  }
  red[t] = lsum; __syncthreads();
  for (int s = 64; s > 0; s >>= 1) { if (t < s) red[t] += red[t + s]; __syncthreads(); }
  if (t == 0) s_mean = red[0] / (float)kD;
  __syncthreads();
  float m = s_mean;
  float lvar = 0.f;
#pragma unroll
  for (int j = 0; j < 4; j++) { float dv = v[j] - m; lvar += dv * dv; }
  red[t] = lvar; __syncthreads();
  for (int s = 64; s > 0; s >>= 1) { if (t < s) red[t] += red[t + s]; __syncthreads(); }
  if (t == 0) s_rstd = rsqrtf(red[0] / (float)kD + 1e-5f);
  __syncthreads();
  float rs = s_rstd;
#pragma unroll
  for (int j = 0; j < 4; j++) {
    int d = t + j * 128;
    float y = (v[j] - m) * rs * gamma[d] + beta[d];
    if (outf) outf[base + d] = y;
    if (outb) outb[base + d] = f2bf_bits(y);
    if (accmode == 1) accum[base + d] = y;
    else if (accmode == 2) accum[base + d] += y;
  }
}

// Final LN with (B,S,D) -> (S,B,D) transposed store into d_out
__global__ __launch_bounds__(128)
void final_ln_kernel(const float* __restrict__ x, const float* __restrict__ gamma,
                     const float* __restrict__ beta, float* __restrict__ out) {
  __shared__ float red[128];
  __shared__ float s_mean, s_rstd;
  const int row = blockIdx.x;              // b*S + s
  const int t = threadIdx.x;
  const size_t base = (size_t)row * kD;
  float v[4];
  float lsum = 0.f;
#pragma unroll
  for (int j = 0; j < 4; j++) { int d = t + j * 128; float a = x[base + d]; v[j] = a; lsum += a; }
  red[t] = lsum; __syncthreads();
  for (int s = 64; s > 0; s >>= 1) { if (t < s) red[t] += red[t + s]; __syncthreads(); }
  if (t == 0) s_mean = red[0] / (float)kD;
  __syncthreads();
  float m = s_mean;
  float lvar = 0.f;
#pragma unroll
  for (int j = 0; j < 4; j++) { float dv = v[j] - m; lvar += dv * dv; }
  red[t] = lvar; __syncthreads();
  for (int s = 64; s > 0; s >>= 1) { if (t < s) red[t] += red[t + s]; __syncthreads(); }
  if (t == 0) s_rstd = rsqrtf(red[0] / (float)kD + 1e-5f);
  __syncthreads();
  float rs = s_rstd;
  const int bb = row / kS, ss = row % kS;
#pragma unroll
  for (int j = 0; j < 4; j++) {
    int d = t + j * 128;
    out[((size_t)ss * kB + bb) * kD + d] = (v[j] - m) * rs * gamma[d] + beta[d];
  }
}

// ---------------- small elementwise kernels ----------------
__global__ void cvt_kernel(const float* __restrict__ src, unsigned short* __restrict__ dst, int n) {
  int i = blockIdx.x * 256 + threadIdx.x;
  if (i < n) dst[i] = f2bf_bits(src[i]);
}

__global__ void tin_kernel(const float* __restrict__ in, float* __restrict__ xf,
                           unsigned short* __restrict__ xb) {
  size_t i = (size_t)blockIdx.x * 256 + threadIdx.x;   // over S*B*D, in-order of input
  int d = (int)(i % kD);
  size_t sb = i / kD;
  int bb = (int)(sb % kB);
  int ss = (int)(sb / kB);
  float v = in[i];
  size_t o = ((size_t)bb * kS + ss) * kD + d;
  xf[o] = v; xb[o] = f2bf_bits(v);
}

__global__ void copy_kernel(const float* __restrict__ src, float* __restrict__ df,
                            unsigned short* __restrict__ db, int n) {
  int i = blockIdx.x * 256 + threadIdx.x;
  if (i < n) { float v = src[i]; df[i] = v; db[i] = f2bf_bits(v); }
}

// =====================================================================
// Host orchestration
// =====================================================================
extern "C" void kernel_launch(void* const* d_in, const int* in_sizes, int n_in,
                              void* d_out, int out_size, void* d_ws, size_t ws_size,
                              hipStream_t stream) {
  (void)in_sizes; (void)n_in; (void)out_size; (void)ws_size;

  const float*         inputs = (const float*)d_in[0];
  const unsigned char* kpm    = (const unsigned char*)d_in[1];
  const int*           spk    = (const int*)d_in[2];
  const float* Wqkv = (const float*)d_in[3];
  const float* bqkv = (const float*)d_in[4];
  const float* Wo   = (const float*)d_in[5];
  const float* bo   = (const float*)d_in[6];
  const float* g1   = (const float*)d_in[7];
  const float* be1  = (const float*)d_in[8];
  const float* W1   = (const float*)d_in[9];
  const float* bf1  = (const float*)d_in[10];
  const float* W2   = (const float*)d_in[11];
  const float* bf2  = (const float*)d_in[12];
  const float* g2   = (const float*)d_in[13];
  const float* be2  = (const float*)d_in[14];
  const float* gN   = (const float*)d_in[15];
  const float* bN   = (const float*)d_in[16];

  // ---- workspace arena ----
  unsigned long long p = (unsigned long long)d_ws;
  auto carve = [&](size_t bytes) -> void* {
    p = (p + 255ull) & ~255ull;
    void* r = (void*)p; p += bytes; return r;
  };
  const size_t nWqkv = (size_t)kNE * kL * 3 * kD * kD;
  const size_t nWo   = (size_t)kNE * kL * kD * kD;
  const size_t nW1   = (size_t)kNE * kL * kFF * kD;
  const size_t nW2   = (size_t)kNE * kL * kD * kFF;
  const size_t nX    = (size_t)kM * kD;
  const size_t nQKV  = (size_t)kM * 3 * kD;
  const size_t nH1   = (size_t)kM * kFF;

  unsigned short* wqkv_bf = (unsigned short*)carve(nWqkv * 2);
  unsigned short* wo_bf   = (unsigned short*)carve(nWo * 2);
  unsigned short* w1_bf   = (unsigned short*)carve(nW1 * 2);
  unsigned short* w2_bf   = (unsigned short*)carve(nW2 * 2);
  float*          x_f32   = (float*)carve(nX * 4);
  unsigned short* x_bf    = (unsigned short*)carve(nX * 2);
  unsigned short* qkv_bf  = (unsigned short*)carve(nQKV * 2);
  unsigned short* o_bf    = (unsigned short*)carve(nX * 2);
  float*          proj    = (float*)carve(nX * 4);
  float*          x1_f32  = (float*)carve(nX * 4);
  unsigned short* x1_bf   = (unsigned short*)carve(nX * 2);
  unsigned short* h1_bf   = (unsigned short*)carve(nH1 * 2);
  float*          ffn     = (float*)carve(nX * 4);
  float*          acc     = (float*)carve(nX * 4);

  // allow >64KB dynamic LDS for the fused attention kernel (CDNA5: 320KB/WGP)
  (void)hipFuncSetAttribute((const void*)attn_kernel,
                            hipFuncAttributeMaxDynamicSharedMemorySize, kATTN_SH);

  // ---- one-time conversions (re-done every call; deterministic) ----
  auto cvt = [&](const float* s, unsigned short* d, size_t n) {
    cvt_kernel<<<(unsigned)((n + 255) / 256), 256, 0, stream>>>(s, d, (int)n);
  };
  cvt(Wqkv, wqkv_bf, nWqkv);
  cvt(Wo,   wo_bf,   nWo);
  cvt(W1,   w1_bf,   nW1);
  cvt(W2,   w2_bf,   nW2);
  tin_kernel<<<(unsigned)(nX / 256), 256, 0, stream>>>(inputs, x_f32, x_bf);

  auto gemm = [&](const unsigned short* A, const unsigned short* Wt, const float* bias,
                  unsigned short* ob, float* of, int N, int K, int mode) {
    dim3 grid(N / kBN, kM / kBM);
    gemm_bf16_kernel<<<grid, 256, 0, stream>>>(A, Wt, bias, ob, of, kM, N, K, mode);
  };

  for (int l = 0; l < kL; l++) {
    for (int e = 0; e < kNE; e++) {
      const int el = e * kL + l;
      // QKV projection: (8192,512) x (1536,512)^T -> bf16 qkv
      gemm(x_bf, wqkv_bf + (size_t)el * 3 * kD * kD, bqkv + (size_t)el * 3 * kD,
           qkv_bf, nullptr, 3 * kD, kD, 0);
      // fused masked attention
      attn_kernel<<<kB * kH * (kS / 32), 256, kATTN_SH, stream>>>(
          qkv_bf, spk, kpm, o_bf, e);
      // output projection -> fp32
      gemm(o_bf, wo_bf + (size_t)el * kD * kD, bo + (size_t)el * kD,
           nullptr, proj, kD, kD, 2);
      // x1 = LN(x + proj)
      ln_kernel<<<kM, 128, 0, stream>>>(x_f32, proj,
          g1 + (size_t)el * kD, be1 + (size_t)el * kD, x1_f32, x1_bf, nullptr, 0);
      // h1 = relu(x1 @ W1^T + b) -> bf16
      gemm(x1_bf, w1_bf + (size_t)el * kFF * kD, bf1 + (size_t)el * kFF,
           h1_bf, nullptr, kFF, kD, 1);
      // ffn = h1 @ W2^T + b -> fp32
      gemm(h1_bf, w2_bf + (size_t)el * kD * kFF, bf2 + (size_t)el * kD,
           nullptr, ffn, kD, kFF, 2);
      // acc (=|+=) LN(x1 + ffn)
      ln_kernel<<<kM, 128, 0, stream>>>(x1_f32, ffn,
          g2 + (size_t)el * kD, be2 + (size_t)el * kD, nullptr, nullptr,
          acc, (e == 0) ? 1 : 2);
    }
    // x <- sum of experts (fp32 + bf16 copies)
    copy_kernel<<<(unsigned)(nX / 256), 256, 0, stream>>>(acc, x_f32, x_bf, (int)nX);
  }

  // final LN with transposed (S,B,D) store
  final_ln_kernel<<<kM, 128, 0, stream>>>(x_f32, gN, bN, (float*)d_out);
}
